// Attention_14740327760418
// MI455X (gfx1250) — matrix operand order repaired and checked by reading, not compiled, not run on hardware
//
#include <hip/hip_runtime.h>

#define BATCH 4
#define SEQ   1024
#define DEMB  256
#define HEADS 8
#define GAMMA 1.0f
#define LN_EPS 1e-5f
#define CONCAT (HEADS*DEMB)   // 2048
#define LDSLD 40              // padded row stride (halves) for the 16x32 P tile

typedef __attribute__((ext_vector_type(16))) __bf16 v16bf;
typedef __attribute__((ext_vector_type(8)))  float  v8f;

union FragAB { v16bf v; int4 q[2]; };

__device__ __forceinline__ unsigned int f2bf(float f) {
  unsigned int u = __float_as_uint(f);
  u += 0x7FFFu + ((u >> 16) & 1u);           // round-to-nearest-even
  return u >> 16;
}

__device__ __forceinline__ int4 pack_bf8(v8f a) {
  unsigned x0 = f2bf(a[0]) | (f2bf(a[1]) << 16);
  unsigned x1 = f2bf(a[2]) | (f2bf(a[3]) << 16);
  unsigned x2 = f2bf(a[4]) | (f2bf(a[5]) << 16);
  unsigned x3 = f2bf(a[6]) | (f2bf(a[7]) << 16);
  int4 r; r.x = (int)x0; r.y = (int)x1; r.z = (int)x2; r.w = (int)x3; return r;
}

__device__ __forceinline__ v8f wmma_bf16(v16bf a, v16bf b, v8f c) {
  return __builtin_amdgcn_wmma_f32_16x16x32_bf16(false, a, false, b, (short)0, c, false, false);
}

// ---------------- LayerNorm (bias-free) -> bf16 ----------------
__global__ void ln_kernel(const float* __restrict__ x, const float* __restrict__ w,
                          unsigned short* __restrict__ xnbf) {
  int row = blockIdx.x;              // B*S rows
  int tid = threadIdx.x;             // 256 threads == DEMB
  float v = x[(size_t)row * DEMB + tid];
  __shared__ float sbuf[DEMB];
  sbuf[tid] = v;
  __syncthreads();
  for (int st = 128; st > 0; st >>= 1) { if (tid < st) sbuf[tid] += sbuf[tid + st]; __syncthreads(); }
  float mu = sbuf[0] * (1.0f / DEMB);
  __syncthreads();
  float d = v - mu;
  sbuf[tid] = d * d;
  __syncthreads();
  for (int st = 128; st > 0; st >>= 1) { if (tid < st) sbuf[tid] += sbuf[tid + st]; __syncthreads(); }
  float var = sbuf[0] * (1.0f / DEMB);
  float r = rsqrtf(var + LN_EPS);
  xnbf[(size_t)row * DEMB + tid] = (unsigned short)f2bf(d * r * w[tid]);
}

// -------- transpose f32 [batch][rows][cols] -> bf16 [batch][cols][rows] --------
__global__ void transpose_bf16_kernel(const float* __restrict__ in, unsigned short* __restrict__ out,
                                      int batch, int rows, int cols) {
  long long idx = (long long)blockIdx.x * blockDim.x + threadIdx.x;
  long long total = (long long)batch * rows * cols;
  if (idx >= total) return;
  int c = (int)(idx % cols);
  long long t = idx / cols;
  int r = (int)(t % rows);
  int bi = (int)(t / rows);
  out[(long long)bi * rows * cols + (long long)c * rows + r] = (unsigned short)f2bf(in[idx]);
}

// ---------------- Q/K projection: C(M=e,N=s) = Wt[e][:] . xn[s][:] ----------------
// A = Wt rows (e-major), B^T = xn rows; both contiguous. Stores pack b128.
__global__ void proj_qk_kernel(const unsigned short* __restrict__ xnbf,
                               const unsigned short* __restrict__ wqt,
                               const unsigned short* __restrict__ wkt,
                               unsigned short* __restrict__ Q,
                               unsigned short* __restrict__ K) {
  int wave = blockIdx.x * (blockDim.x >> 5) + (threadIdx.x >> 5);
  int lane = threadIdx.x & 31, lo = lane & 15, hi = lane >> 4;
  int sstrip = wave & 15; wave >>= 4;
  int estrip = wave & 15; wave >>= 4;
  int h = wave & 7; int b = wave >> 3;
  int e0 = estrip * 16, s0 = sstrip * 64;
  const unsigned short* xb = xnbf + (size_t)b * SEQ * DEMB;
  const unsigned short* wq = wqt + (size_t)h * DEMB * DEMB;
  const unsigned short* wk = wkt + (size_t)h * DEMB * DEMB;
  v8f accq[4] = {}; v8f acck[4] = {};
  for (int c = 0; c < 8; ++c) {
    int d0 = c * 32;
    FragAB aq, ak;
    const unsigned short* qa = wq + (size_t)(e0 + lo) * DEMB + d0 + 16 * hi;
    const unsigned short* ka = wk + (size_t)(e0 + lo) * DEMB + d0 + 16 * hi;
    aq.q[0] = *(const int4*)qa; aq.q[1] = *(const int4*)(qa + 8);
    ak.q[0] = *(const int4*)ka; ak.q[1] = *(const int4*)(ka + 8);
#pragma unroll
    for (int j = 0; j < 4; ++j) {
      FragAB bx;
      const unsigned short* xr = xb + (size_t)(s0 + 16 * j + lo) * DEMB + d0 + 16 * hi;
      bx.q[0] = *(const int4*)xr; bx.q[1] = *(const int4*)(xr + 8);
      accq[j] = wmma_bf16(aq.v, bx.v, accq[j]);
      acck[j] = wmma_bf16(ak.v, bx.v, acck[j]);
    }
  }
  size_t qbase = (size_t)(b * HEADS + h) * SEQ;
#pragma unroll
  for (int j = 0; j < 4; ++j) {
    int srow = s0 + 16 * j + lo;
    *(int4*)(Q + (qbase + srow) * DEMB + e0 + 8 * hi) = pack_bf8(accq[j]);
    *(int4*)(K + (qbase + srow) * DEMB + e0 + 8 * hi) = pack_bf8(acck[j]);
  }
}

// ---------------- V projection: C(M=s,N=e) -> store transposed Vt[d][s] ----------------
__global__ void proj_v_kernel(const unsigned short* __restrict__ xnbf,
                              const unsigned short* __restrict__ wvt,
                              unsigned short* __restrict__ Vt) {
  int wave = blockIdx.x * (blockDim.x >> 5) + (threadIdx.x >> 5);
  int lane = threadIdx.x & 31, lo = lane & 15, hi = lane >> 4;
  int estrip = wave & 3;  wave >>= 2;    // 4 strips of 64 e
  int sstrip = wave & 63; wave >>= 6;    // 64 strips of 16 s
  int h = wave & 7; int b = wave >> 3;
  int e0 = estrip * 64, s0 = sstrip * 16;
  const unsigned short* xb = xnbf + (size_t)b * SEQ * DEMB;
  const unsigned short* wv = wvt + (size_t)h * DEMB * DEMB;
  v8f acc[4] = {};
  for (int c = 0; c < 8; ++c) {
    int d0 = c * 32;
    FragAB ax;
    const unsigned short* xa = xb + (size_t)(s0 + lo) * DEMB + d0 + 16 * hi;
    ax.q[0] = *(const int4*)xa; ax.q[1] = *(const int4*)(xa + 8);
#pragma unroll
    for (int j = 0; j < 4; ++j) {
      FragAB bw;
      const unsigned short* wr = wv + (size_t)(e0 + 16 * j + lo) * DEMB + d0 + 16 * hi;
      bw.q[0] = *(const int4*)wr; bw.q[1] = *(const int4*)(wr + 8);
      acc[j] = wmma_bf16(ax.v, bw.v, acc[j]);
    }
  }
  size_t vbase = (size_t)(b * HEADS + h) * DEMB;
#pragma unroll
  for (int j = 0; j < 4; ++j) {
    int erow = e0 + 16 * j + lo;
    *(int4*)(Vt + (vbase + erow) * SEQ + s0 + 8 * hi) = pack_bf8(acc[j]);
  }
}

// ---------------- row sum of squares over D (from bf16) ----------------
__global__ void rowsumsq_kernel(const unsigned short* __restrict__ m, float* __restrict__ out, int rows) {
  int row = blockIdx.x * blockDim.x + threadIdx.x;
  if (row >= rows) return;
  const int4* p = (const int4*)(m + (size_t)row * DEMB);
  float s = 0.f;
  for (int i = 0; i < DEMB / 8; ++i) {
    int4 t = p[i];
    unsigned a[4] = {(unsigned)t.x, (unsigned)t.y, (unsigned)t.z, (unsigned)t.w};
#pragma unroll
    for (int j = 0; j < 4; ++j) {
      float flo = __uint_as_float(a[j] << 16);
      float fhi = __uint_as_float(a[j] & 0xFFFF0000u);
      s += flo * flo + fhi * fhi;
    }
  }
  out[row] = s;
}

// ---------------- flash-style RBF attention ----------------
// wave owns 16 queries x full D=256 output (16 C-tiles).
__global__ void __launch_bounds__(128, 1)
rbf_attn_kernel(const unsigned short* __restrict__ Q,
                const unsigned short* __restrict__ K,
                const unsigned short* __restrict__ Vt,
                const float* __restrict__ q2,
                const float* __restrict__ k2,
                unsigned short* __restrict__ concat) {
  __shared__ unsigned short ptile[4][16 * LDSLD];
  int wid = threadIdx.x >> 5;
  int wave = blockIdx.x * 4 + wid;
  int lane = threadIdx.x & 31, lo = lane & 15, hi = lane >> 4;
  int qstrip = wave & 63; wave >>= 6;
  int h = wave & 7; int b = wave >> 3;
  int q0 = qstrip * 16;
  size_t bh = (size_t)(b * HEADS + h);
  const unsigned short* Qb = Q + bh * SEQ * DEMB;
  const unsigned short* Kb = K + bh * SEQ * DEMB;
  const unsigned short* Vb = Vt + bh * DEMB * SEQ;
  const float* q2b = q2 + bh * SEQ;
  const float* k2b = k2 + bh * SEQ;

  float q2v[8];
#pragma unroll
  for (int r = 0; r < 8; ++r) q2v[r] = q2b[q0 + r + 8 * hi];

  v8f o[16] = {};
  unsigned short* pt = &ptile[wid][0];
  const float NEG_G_LOG2E = -GAMMA * 1.4426950408889634f;

  for (int kt = 0; kt < SEQ; kt += 32) {
    v8f s0t = {}; v8f s1t = {};
    for (int c = 0; c < 8; ++c) {
      int d0 = c * 32;
      FragAB aq;
      const unsigned short* qr = Qb + (size_t)(q0 + lo) * DEMB + d0 + 16 * hi;
      aq.q[0] = *(const int4*)qr; aq.q[1] = *(const int4*)(qr + 8);
      FragAB bk0, bk1;
      const unsigned short* kr0 = Kb + (size_t)(kt + lo) * DEMB + d0 + 16 * hi;
      const unsigned short* kr1 = Kb + (size_t)(kt + 16 + lo) * DEMB + d0 + 16 * hi;
      bk0.q[0] = *(const int4*)kr0; bk0.q[1] = *(const int4*)(kr0 + 8);
      bk1.q[0] = *(const int4*)kr1; bk1.q[1] = *(const int4*)(kr1 + 8);
      s0t = wmma_bf16(aq.v, bk0.v, s0t);
      s1t = wmma_bf16(aq.v, bk1.v, s1t);
    }
    float k2a = k2b[kt + lo];
    float k2c = k2b[kt + 16 + lo];
#pragma unroll
    for (int r = 0; r < 8; ++r) {
      float d0v = fmaxf(q2v[r] + k2a - 2.0f * s0t[r], 0.0f);
      float d1v = fmaxf(q2v[r] + k2c - 2.0f * s1t[r], 0.0f);
      float p0 = exp2f(NEG_G_LOG2E * d0v);
      float p1 = exp2f(NEG_G_LOG2E * d1v);
      pt[(r + 8 * hi) * LDSLD + lo]      = (unsigned short)f2bf(p0);
      pt[(r + 8 * hi) * LDSLD + 16 + lo] = (unsigned short)f2bf(p1);
    }
    __syncthreads();
    FragAB ap;
    const unsigned short* pr = pt + (size_t)lo * LDSLD + 16 * hi;
    ap.q[0] = *(const int4*)pr; ap.q[1] = *(const int4*)(pr + 8);
#pragma unroll
    for (int n = 0; n < 16; ++n) {
      FragAB bv;
      const unsigned short* vr = Vb + (size_t)(n * 16 + lo) * SEQ + kt + 16 * hi;
      bv.q[0] = *(const int4*)vr; bv.q[1] = *(const int4*)(vr + 8);
      o[n] = wmma_bf16(ap.v, bv.v, o[n]);
    }
    __syncthreads();
  }

  unsigned short* ob = concat + (size_t)b * SEQ * CONCAT + (size_t)h * DEMB;
#pragma unroll
  for (int n = 0; n < 16; ++n)
#pragma unroll
    for (int r = 0; r < 8; ++r)
      ob[(size_t)(q0 + r + 8 * hi) * CONCAT + n * 16 + lo] = (unsigned short)f2bf(o[n][r]);
}

// ---------------- output projection: C(M=e_out,N=s) = Wot[e][:2048] . concat[s][:2048] ----------------
__global__ void outproj_kernel(const unsigned short* __restrict__ concat,
                               const unsigned short* __restrict__ wot,
                               float* __restrict__ out) {
  int wave = blockIdx.x * (blockDim.x >> 5) + (threadIdx.x >> 5);
  int lane = threadIdx.x & 31, lo = lane & 15, hi = lane >> 4;
  int sstrip = wave & 15; wave >>= 4;
  int estrip = wave & 15; wave >>= 4;
  int b = wave;
  int e0 = estrip * 16, s0 = sstrip * 64;
  const unsigned short* cb = concat + (size_t)b * SEQ * CONCAT;
  v8f acc[4] = {};
  for (int c = 0; c < 64; ++c) {
    int d0 = c * 32;
    FragAB aw;
    const unsigned short* ar = wot + (size_t)(e0 + lo) * CONCAT + d0 + 16 * hi;
    aw.q[0] = *(const int4*)ar; aw.q[1] = *(const int4*)(ar + 8);
#pragma unroll
    for (int j = 0; j < 4; ++j) {
      FragAB bx;
      const unsigned short* xr = cb + (size_t)(s0 + 16 * j + lo) * CONCAT + d0 + 16 * hi;
      bx.q[0] = *(const int4*)xr; bx.q[1] = *(const int4*)(xr + 8);
      acc[j] = wmma_bf16(aw.v, bx.v, acc[j]);
    }
  }
#pragma unroll
  for (int j = 0; j < 4; ++j) {
    float* orow = out + (size_t)(b * SEQ + s0 + 16 * j + lo) * DEMB + e0 + 8 * hi;
    float4 f0 = make_float4(acc[j][0], acc[j][1], acc[j][2], acc[j][3]);
    float4 f1 = make_float4(acc[j][4], acc[j][5], acc[j][6], acc[j][7]);
    *(float4*)orow = f0;
    *(float4*)(orow + 4) = f1;
  }
}

extern "C" void kernel_launch(void* const* d_in, const int* in_sizes, int n_in,
                              void* d_out, int out_size, void* d_ws, size_t ws_size,
                              hipStream_t stream) {
  const float* x   = (const float*)d_in[0];
  const float* Wq  = (const float*)d_in[3];
  const float* Wk  = (const float*)d_in[4];
  const float* Wv  = (const float*)d_in[5];
  const float* Wo  = (const float*)d_in[6];
  const float* lnw = (const float*)d_in[7];
  float* out = (float*)d_out;

  char* ws = (char*)d_ws;
  size_t off = 0;
  auto alloc = [&](size_t bytes) -> void* {
    void* p = ws + off;
    off += (bytes + 255) & ~(size_t)255;
    return p;
  };
  const size_t szXn  = (size_t)BATCH * SEQ * DEMB * 2;
  const size_t szW   = (size_t)HEADS * DEMB * DEMB * 2;
  const size_t szWo  = (size_t)CONCAT * DEMB * 2;
  const size_t szQKV = (size_t)BATCH * HEADS * SEQ * DEMB * 2;
  const size_t szS   = (size_t)BATCH * HEADS * SEQ * 4;
  const size_t szCat = (size_t)BATCH * SEQ * CONCAT * 2;

  unsigned short* xnbf = (unsigned short*)alloc(szXn);
  unsigned short* wqt  = (unsigned short*)alloc(szW);
  unsigned short* wkt  = (unsigned short*)alloc(szW);
  unsigned short* wvt  = (unsigned short*)alloc(szW);
  unsigned short* wot  = (unsigned short*)alloc(szWo);
  unsigned short* Qb   = (unsigned short*)alloc(szQKV);
  unsigned short* Kb   = (unsigned short*)alloc(szQKV);
  unsigned short* Vtb  = (unsigned short*)alloc(szQKV);
  float*          q2   = (float*)alloc(szS);
  float*          k2   = (float*)alloc(szS);
  unsigned short* cat  = (unsigned short*)alloc(szCat);

  // 1) LayerNorm -> bf16
  ln_kernel<<<BATCH * SEQ, DEMB, 0, stream>>>(x, lnw, xnbf);

  // 2) weight transposes -> bf16 (e-major rows)
  int tW = HEADS * DEMB * DEMB;      // 524288
  transpose_bf16_kernel<<<(tW + 255) / 256, 256, 0, stream>>>(Wq, wqt, HEADS, DEMB, DEMB);
  transpose_bf16_kernel<<<(tW + 255) / 256, 256, 0, stream>>>(Wk, wkt, HEADS, DEMB, DEMB);
  transpose_bf16_kernel<<<(tW + 255) / 256, 256, 0, stream>>>(Wv, wvt, HEADS, DEMB, DEMB);
  int tWo = CONCAT * DEMB;           // 524288
  transpose_bf16_kernel<<<(tWo + 255) / 256, 256, 0, stream>>>(Wo, wot, 1, CONCAT, DEMB);

  // 3) projections (WMMA)
  proj_qk_kernel<<<512, 256, 0, stream>>>(xnbf, wqt, wkt, Qb, Kb);   // 4096 waves
  proj_v_kernel<<<1024, 256, 0, stream>>>(xnbf, wvt, Vtb);           // 8192 waves

  // 4) ||q||^2, ||k||^2
  int rows = BATCH * HEADS * SEQ;    // 32768
  rowsumsq_kernel<<<rows / 256, 256, 0, stream>>>(Qb, q2, rows);
  rowsumsq_kernel<<<rows / 256, 256, 0, stream>>>(Kb, k2, rows);

  // 5) flash-style RBF attention (WMMA QK^T + exp + WMMA PV)
  rbf_attn_kernel<<<512, 128, 0, stream>>>(Qb, Kb, Vtb, q2, k2, cat); // 2048 waves

  // 6) output projection (WMMA)
  outproj_kernel<<<128, 256, 0, stream>>>(cat, wot, out);             // 1024 waves
}
